// FactorizedCrossAttention_63556926046900
// MI455X (gfx1250) — matrix operand order repaired and checked by reading, not compiled, not run on hardware
//
#include <hip/hip_runtime.h>
#include <hip/hip_bf16.h>

typedef __bf16 bf16_t;
typedef __attribute__((ext_vector_type(16))) __bf16 v16bf;
typedef __attribute__((ext_vector_type(8)))  __bf16 v8bf;
typedef __attribute__((ext_vector_type(4)))  __bf16 v4bf;
typedef __attribute__((ext_vector_type(8)))  float  v8f;

#define D_MODEL 1024
#define NUM_HEADS 16
#define HEAD_DIM 64
#define QUERY_GROUPS 4
#define T_TOKENS 77
#define TKPAD 96           // keys padded to multiple of 32 for the PV WMMA
#define B_SZ 2
#define T_FRAMES 16
#define HW_SZ 1024
#define NQ (B_SZ * T_FRAMES * HW_SZ)   // 32768 query rows

// ---------------------------------------------------------------- helpers
__device__ __forceinline__ v16bf make_frag(v8bf lo, v8bf hi) {
    v16bf r;
#pragma unroll
    for (int i = 0; i < 8; ++i) { r[i] = lo[i]; r[i + 8] = hi[i]; }
    return r;
}
__device__ __forceinline__ void store_out(float* p, float v)  { *p = v; }
__device__ __forceinline__ void store_out(bf16_t* p, float v) { *p = (bf16_t)v; }

// async copy of 16B (per lane) global -> LDS, tracked by ASYNCcnt
__device__ __forceinline__ void async_b128(const bf16_t* gp, bf16_t* lp) {
    unsigned lds = (unsigned)(uintptr_t)lp;                     // LDS byte offset = addr[31:0]
    unsigned long long ga = (unsigned long long)(uintptr_t)gp;
    asm volatile("global_load_async_to_lds_b128 %0, %1, off"
                 :: "v"(lds), "v"(ga) : "memory");
}
__device__ __forceinline__ void wait_async0() {
    asm volatile("s_wait_asynccnt 0x0" ::: "memory");
}

// ---------------------------------------------------------------- elementwise
// float4 -> 4x bf16 (n must be a multiple of 4; all call sites are)
__global__ void cvt_kernel(const float* __restrict__ src, bf16_t* __restrict__ dst, int n4) {
    int i = blockIdx.x * blockDim.x + threadIdx.x;
    if (i < n4) {
        float4 f = ((const float4*)src)[i];
        v4bf o;
        o[0] = (bf16_t)f.x; o[1] = (bf16_t)f.y; o[2] = (bf16_t)f.z; o[3] = (bf16_t)f.w;
        ((v4bf*)dst)[i] = o;
    }
}
// dst[i] = bf16(src[i] + src[i+off])   (Wst top + bottom fold)
__global__ void addcvt_kernel(const float* __restrict__ src, bf16_t* __restrict__ dst,
                              int n, int off) {
    int i = blockIdx.x * blockDim.x + threadIdx.x;
    if (i < n) dst[i] = (bf16_t)(src[i] + src[i + off]);
}

// ---------------------------------------------------------------- K/V projection
// K stored [B][G][TKPAD][64] (row-major keys, softmax scale folded in);
// V stored transposed [B][G][64][TKPAD] so WMMA B-fragments are contiguous.
__global__ void kv_kernel(const float* __restrict__ text, const float* __restrict__ Wk,
                          const float* __restrict__ Wv, bf16_t* __restrict__ Kc,
                          bf16_t* __restrict__ Vt) {
    int idx = blockIdx.x * blockDim.x + threadIdx.x;
    if (idx >= B_SZ * QUERY_GROUPS * TKPAD * HEAD_DIM) return;
    int d   = idx & (HEAD_DIM - 1);
    int tok = (idx >> 6) % TKPAD;
    int g   = ((idx >> 6) / TKPAD) % QUERY_GROUPS;
    int b   = idx / (QUERY_GROUPS * TKPAD * HEAD_DIM);
    float kv = 0.f, vv = 0.f;
    if (tok < T_TOKENS) {
        const float* trow = text + ((size_t)b * T_TOKENS + tok) * D_MODEL;
        int wcol = g * HEAD_DIM + d;
        for (int c = 0; c < D_MODEL; ++c) {
            float t = trow[c];
            kv += t * Wk[(size_t)c * (QUERY_GROUPS * HEAD_DIM) + wcol];
            vv += t * Wv[(size_t)c * (QUERY_GROUPS * HEAD_DIM) + wcol];
        }
    }
    Kc[((size_t)(b * QUERY_GROUPS + g) * TKPAD + tok) * HEAD_DIM + d] = (bf16_t)(kv * 0.125f);
    Vt[((size_t)(b * QUERY_GROUPS + g) * HEAD_DIM + d) * TKPAD + tok] = (bf16_t)vv;
}

// ---------------------------------------------------------------- WMMA GEMM
// C[M,N] = A[M,K] * B[K,N], A/B bf16 row-major, f32 accumulate.
// 128x128x32 block tile, 8 wave32 waves, each wave owns 64x32 (4x2 16x16 tiles).
// Double-buffered LDS; A tile staged with GLOBAL_LOAD_ASYNC_TO_LDS_B128,
// B tile staged through registers (transposed in LDS), one barrier per K-step.
#define BM 128
#define BN 128
#define BK 32
#define LDA 40   // bf16 stride (80B, 16B aligned, conflict-padded)
#define LDB 40

template <typename OutT>
__global__ __launch_bounds__(256) void gemm_bf16(const bf16_t* __restrict__ A,
                                                 const bf16_t* __restrict__ B,
                                                 OutT* __restrict__ C,
                                                 int M, int N, int K) {
    __shared__ bf16_t As[2][BM * LDA];
    __shared__ bf16_t Bs[2][BN * LDB];   // transposed: Bs[buf][n*LDB + k]
    int tid = threadIdx.x;
    int lane = tid & 31, wid = tid >> 5;
    int wm = wid >> 2, wn = wid & 3;
    int l16 = lane & 15, hf = lane >> 4;
    int bm = blockIdx.y * BM, bn = blockIdx.x * BN;

    v8f acc[4][2];
#pragma unroll
    for (int i = 0; i < 4; ++i)
#pragma unroll
        for (int j = 0; j < 2; ++j)
            acc[i][j] = (v8f){0.f, 0.f, 0.f, 0.f, 0.f, 0.f, 0.f, 0.f};

    int arow = tid >> 1, acol = (tid & 1) << 4;      // A: 32B per thread
    int brow = tid >> 3, bcol = (tid & 7) << 4;      // B: 32B per thread

    const bf16_t* aptr = A + (size_t)(bm + arow) * K + acol;
    const bf16_t* bptr = B + (size_t)brow * N + bn + bcol;
    bf16_t* asl = &As[0][arow * LDA + acol];         // LDS dest (buffer 0)
    const unsigned abufstride = (unsigned)(BM * LDA);

    // ---- prologue: stage tile 0
    async_b128(aptr,     asl);
    async_b128(aptr + 8, asl + 8);
    {
        v8bf b0 = *(const v8bf*)bptr;
        v8bf b1 = *(const v8bf*)(bptr + 8);
#pragma unroll
        for (int j = 0; j < 8; ++j) {
            Bs[0][(bcol + j) * LDB + brow]     = b0[j];
            Bs[0][(bcol + 8 + j) * LDB + brow] = b1[j];
        }
    }
    wait_async0();
    __syncthreads();

    const int nsteps = K / BK;
    for (int s = 0; s < nsteps; ++s) {
        int cur = s & 1, nxt = cur ^ 1;
        v8bf nb0 = {}, nb1 = {};
        bool more = (s + 1 < nsteps);
        if (more) {       // start next-slab movement before computing
            const bf16_t* an = aptr + (size_t)(s + 1) * BK;
            bf16_t* al = asl + (nxt ? abufstride : 0u) - (cur ? abufstride : 0u);
            // simpler: recompute dest
            al = &As[nxt][arow * LDA + acol];
            async_b128(an,     al);
            async_b128(an + 8, al + 8);
            const bf16_t* bn_ = bptr + (size_t)(s + 1) * BK * N;
            nb0 = *(const v8bf*)bn_;
            nb1 = *(const v8bf*)(bn_ + 8);
            __builtin_prefetch(an + BK, 0, 0);        // global_prefetch_b8
        }

        // ---- compute on buffer `cur`
        v16bf afrag[4], bfrag[2];
#pragma unroll
        for (int i = 0; i < 4; ++i) {                 // A frag: lanes<16 K:0-7,16-23
            int row = wm * 64 + i * 16 + l16;
            int kk = hf ? 8 : 0;
            v8bf lo = *(const v8bf*)(&As[cur][row * LDA + kk]);
            v8bf hi = *(const v8bf*)(&As[cur][row * LDA + kk + 16]);
            afrag[i] = make_frag(lo, hi);
        }
#pragma unroll
        for (int j = 0; j < 2; ++j) {                 // B frag: lanes<16 K:0-7,16-23
            int col = wn * 32 + j * 16 + l16;
            int kk = hf ? 8 : 0;
            v8bf lo = *(const v8bf*)(&Bs[cur][col * LDB + kk]);
            v8bf hi = *(const v8bf*)(&Bs[cur][col * LDB + kk + 16]);
            bfrag[j] = make_frag(lo, hi);
        }
#pragma unroll
        for (int i = 0; i < 4; ++i)
#pragma unroll
            for (int j = 0; j < 2; ++j)
                acc[i][j] = __builtin_amdgcn_wmma_f32_16x16x32_bf16(
                    false, afrag[i], false, bfrag[j], (short)0, acc[i][j], false, false);

        if (more) {       // finish staging buffer `nxt`
#pragma unroll
            for (int j = 0; j < 8; ++j) {
                Bs[nxt][(bcol + j) * LDB + brow]     = nb0[j];
                Bs[nxt][(bcol + 8 + j) * LDB + brow] = nb1[j];
            }
            wait_async0();
        }
        __syncthreads();
    }

#pragma unroll
    for (int i = 0; i < 4; ++i)
#pragma unroll
        for (int j = 0; j < 2; ++j)
#pragma unroll
            for (int r = 0; r < 8; ++r) {
                int row = bm + wm * 64 + i * 16 + r + hf * 8;
                int col = bn + wn * 32 + j * 16 + l16;
                store_out(C + (size_t)row * N + col, acc[i][j][r]);
            }
}

// ---------------------------------------------------------------- attention
// One wave per (16-query tile, head). Scores via WMMA, masked softmax,
// P staged in LDS (A-fragment order), PV via WMMA against transposed V.
#define PSTR 104   // 16B-aligned padded row stride for P in LDS

__global__ __launch_bounds__(128) void attn_kernel(const bf16_t* __restrict__ Q,
                                                   const bf16_t* __restrict__ Kc,
                                                   const bf16_t* __restrict__ Vt,
                                                   const int* __restrict__ pmask,
                                                   bf16_t* __restrict__ O) {
    __shared__ bf16_t P[4][16 * PSTR];
    int tid = threadIdx.x;
    int lane = tid & 31, wid = tid >> 5;
    int l16 = lane & 15, hf = lane >> 4;

    int tileid = blockIdx.x * 4 + wid;
    int head = tileid & (NUM_HEADS - 1);
    int r0 = (tileid >> 4) * 16;
    int b = r0 / (T_FRAMES * HW_SZ);
    int g = head >> 2;      // HPG = 4

    const bf16_t* kbase  = Kc + (size_t)(b * QUERY_GROUPS + g) * TKPAD * HEAD_DIM;
    const bf16_t* vtbase = Vt + (size_t)(b * QUERY_GROUPS + g) * HEAD_DIM * TKPAD;
    bf16_t* Pw = P[wid];

    // Q fragments (head_dim 64 -> two K=32 chunks)
    v16bf qf[2];
    {
        const bf16_t* qrow = Q + (size_t)(r0 + l16) * D_MODEL + head * HEAD_DIM;
#pragma unroll
        for (int c = 0; c < 2; ++c) {
            int kk = c * 32 + (hf ? 8 : 0);
            qf[c] = make_frag(*(const v8bf*)(qrow + kk), *(const v8bf*)(qrow + kk + 16));
        }
    }
    // scores: 5 key tiles of 16 (77 keys padded to 80)
    v8f sc[5];
#pragma unroll
    for (int nt = 0; nt < 5; ++nt) {
        v8f a = (v8f){0.f, 0.f, 0.f, 0.f, 0.f, 0.f, 0.f, 0.f};
#pragma unroll
        for (int c = 0; c < 2; ++c) {
            const bf16_t* krow = kbase + (size_t)(nt * 16 + l16) * HEAD_DIM
                               + c * 32 + (hf ? 8 : 0);
            v16bf bfr = make_frag(*(const v8bf*)krow, *(const v8bf*)(krow + 16));
            a = __builtin_amdgcn_wmma_f32_16x16x32_bf16(false, qf[c], false, bfr,
                                                        (short)0, a, false, false);
        }
        sc[nt] = a;
    }
    const float NEG = -3.0e38f;
#pragma unroll
    for (int nt = 0; nt < 5; ++nt) {
        int key = nt * 16 + l16;
        bool valid = (key < T_TOKENS) && (pmask[b * T_TOKENS + key] != 0);
        if (!valid) {
#pragma unroll
            for (int r = 0; r < 8; ++r) sc[nt][r] = NEG;
        }
    }
    // softmax over keys; rows live across 16-lane halves (C layout)
    float rsum[8];
#pragma unroll
    for (int r = 0; r < 8; ++r) {
        float m = sc[0][r];
#pragma unroll
        for (int nt = 1; nt < 5; ++nt) m = fmaxf(m, sc[nt][r]);
        for (int s = 1; s < 16; s <<= 1) m = fmaxf(m, __shfl_xor(m, s, 16));
        float sum = 0.f;
#pragma unroll
        for (int nt = 0; nt < 5; ++nt) {
            float p = __expf(sc[nt][r] - m);
            sc[nt][r] = p;
            sum += p;
        }
        for (int s = 1; s < 16; s <<= 1) sum += __shfl_xor(sum, s, 16);
        rsum[r] = sum;
    }
    // stage P in LDS (row-major 16 x TKPAD, cols 80..95 zeroed)
    {
        int cb = 80 + hf * 8;
#pragma unroll
        for (int j = 0; j < 8; ++j) Pw[l16 * PSTR + cb + j] = (bf16_t)0.f;
    }
#pragma unroll
    for (int nt = 0; nt < 5; ++nt)
#pragma unroll
        for (int r = 0; r < 8; ++r)
            Pw[(r + hf * 8) * PSTR + nt * 16 + l16] = (bf16_t)sc[nt][r];

    // PV: P(16x96) x V(96x64)
    v16bf pf[3];
#pragma unroll
    for (int c = 0; c < 3; ++c) {
        int kk = c * 32 + (hf ? 8 : 0);
        pf[c] = make_frag(*(const v8bf*)(Pw + l16 * PSTR + kk),
                          *(const v8bf*)(Pw + l16 * PSTR + kk + 16));
    }
#pragma unroll
    for (int d = 0; d < 4; ++d) {
        v8f a = (v8f){0.f, 0.f, 0.f, 0.f, 0.f, 0.f, 0.f, 0.f};
#pragma unroll
        for (int c = 0; c < 3; ++c) {
            const bf16_t* vrow = vtbase + (size_t)(d * 16 + l16) * TKPAD
                               + c * 32 + (hf ? 8 : 0);
            v16bf bfr = make_frag(*(const v8bf*)vrow, *(const v8bf*)(vrow + 16));
            a = __builtin_amdgcn_wmma_f32_16x16x32_bf16(false, pf[c], false, bfr,
                                                        (short)0, a, false, false);
        }
#pragma unroll
        for (int r = 0; r < 8; ++r) {
            int row = r0 + r + hf * 8;
            int col = head * HEAD_DIM + d * 16 + l16;
            O[(size_t)row * D_MODEL + col] = (bf16_t)(a[r] * (1.0f / rsum[r]));
        }
    }
}

// ---------------------------------------------------------------- launch
extern "C" void kernel_launch(void* const* d_in, const int* in_sizes, int n_in,
                              void* d_out, int out_size, void* d_ws, size_t ws_size,
                              hipStream_t stream) {
    (void)in_sizes; (void)n_in; (void)out_size; (void)ws_size;
    const float* x     = (const float*)d_in[0];
    const float* text  = (const float*)d_in[1];
    const int*   pmask = (const int*)d_in[2];
    const float* Wq    = (const float*)d_in[5];
    const float* Wk    = (const float*)d_in[6];
    const float* Wv    = (const float*)d_in[7];
    const float* Wo    = (const float*)d_in[8];
    const float* Wst   = (const float*)d_in[9];
    float* out = (float*)d_out;

    char* ws = (char*)d_ws;
    size_t off = 0;
    bf16_t* qb    = (bf16_t*)(ws + off); off += (size_t)NQ * D_MODEL * 2;   // Q proj
    bf16_t* xb    = (bf16_t*)(ws + off); off += (size_t)NQ * D_MODEL * 2;   // x bf16, reused for attn
    bf16_t* wqb   = (bf16_t*)(ws + off); off += (size_t)D_MODEL * D_MODEL * 2;
    bf16_t* wsumb = (bf16_t*)(ws + off); off += (size_t)D_MODEL * D_MODEL * 2;
    bf16_t* wob   = (bf16_t*)(ws + off); off += (size_t)D_MODEL * D_MODEL * 2;
    bf16_t* wfb   = (bf16_t*)(ws + off); off += (size_t)D_MODEL * D_MODEL * 2;
    bf16_t* kc    = (bf16_t*)(ws + off); off += (size_t)B_SZ * QUERY_GROUPS * TKPAD * HEAD_DIM * 2;
    bf16_t* vtc   = (bf16_t*)(ws + off); off += (size_t)B_SZ * QUERY_GROUPS * TKPAD * HEAD_DIM * 2;

    const int nX = NQ * D_MODEL;          // 33,554,432
    const int nW = D_MODEL * D_MODEL;     // 1,048,576
    cvt_kernel<<<(nX / 4 + 255) / 256, 256, 0, stream>>>(x, xb, nX / 4);
    cvt_kernel<<<(nW / 4 + 255) / 256, 256, 0, stream>>>(Wq, wqb, nW / 4);
    cvt_kernel<<<(nW / 4 + 255) / 256, 256, 0, stream>>>(Wo, wob, nW / 4);
    addcvt_kernel<<<(nW + 255) / 256, 256, 0, stream>>>(Wst, wsumb, nW, nW);

    const int nKV = B_SZ * QUERY_GROUPS * TKPAD * HEAD_DIM;  // 49152
    kv_kernel<<<(nKV + 255) / 256, 256, 0, stream>>>(text, Wk, Wv, kc, vtc);

    // Q = x @ Wq
    gemm_bf16<bf16_t><<<dim3(D_MODEL / BN, NQ / BM), 256, 0, stream>>>(xb, wqb, qb, NQ, D_MODEL, D_MODEL);
    // W_fused = (Wst_top + Wst_bot) @ Wo
    gemm_bf16<bf16_t><<<dim3(D_MODEL / BN, D_MODEL / BM), 256, 0, stream>>>(wsumb, wob, wfb, D_MODEL, D_MODEL, D_MODEL);
    // attention (writes attn into xb; Q GEMM already consumed it)
    attn_kernel<<<(NQ / 16) * NUM_HEADS / 4, 128, 0, stream>>>(qb, kc, vtc, pmask, xb);
    // out = attn @ W_fused
    gemm_bf16<float><<<dim3(D_MODEL / BN, NQ / BM), 256, 0, stream>>>(xb, wfb, out, NQ, D_MODEL, D_MODEL);
}
